// EarthSpecificBlock_39238821216903
// MI455X (gfx1250) — compile-verified, matrix operand order checked by reading, not run on hardware
//
#include <hip/hip_runtime.h>
#include <hip/hip_bf16.h>

// ---------------------------------------------------------------------------
// EarthSpecificBlock (Pangu 3D shifted-window attention + MLP) for gfx1250.
// All GEMMs use v_wmma_f32_16x16x32_bf16 (wave32), f32 accumulation.
// Pipeline:
//   k0: weights fp32 -> bf16
//   k1: roll + window gather + QKV GEMM (Q pre-scaled, V stored transposed)
//   k2: per (window, head): S=Q@K^T + bias + mask, softmax (register strip),
//       O = P@V  (K padded 144->160).  K slab staged to LDS via TDM.
//   k3a: proj GEMM per window (Xattn staged to LDS via TDM)
//   k3b: window-reverse + unroll + LayerNorm1 + residual -> Yf (f32) / Yb (bf16)
//   k4a: MLP1 + exact GELU -> Hb (bf16)   (4 N-tiles per wave, A reuse)
//   k4b: MLP2 -> Po (f32, reused buffer)  (4 N-tiles per wave, A reuse)
//   k4c: LayerNorm2 + residual -> d_out
// Workspace use ~737 MB (see offsets below).
// ---------------------------------------------------------------------------

#define HEADS 6
#define HD    32
#define Cc    192
#define Zc    8
#define Hc    96
#define Wc    192
#define Twin  144          // tokens per window (2*6*12)
#define WIN   1024         // total windows (16 * 64)
#define NT    147456       // total tokens
#define TWc   64           // "type of windows" = nZ*nH

typedef __attribute__((ext_vector_type(16))) __bf16 v16bf;
typedef __attribute__((ext_vector_type(8)))  float  v8f;

struct FragBF { union { v16bf v; unsigned d[8]; }; };

static __device__ inline unsigned short f2bf(float f) {
  unsigned u = __float_as_uint(f);
  u += 0x7fffu + ((u >> 16) & 1u);            // round-to-nearest-even
  return (unsigned short)(u >> 16);
}

// Load a 16x32 bf16 fragment. Source is "K-contiguous" row-major:
//   element = src[(row0+r)*ld + k], pairs (k,k+1) contiguous -> b128 loads.
// Used for both A (row = M) and B (row = N, data pre-transposed) operands.
static __device__ inline void load_frag(FragBF& f, const unsigned short* base,
                                        int row0, int ld, int kbase) {
  const int lane = threadIdx.x & 31;
  const int rn   = lane & 15;
  const int half = lane >> 4;
  const unsigned short* p = base + (size_t)(row0 + rn) * ld + kbase + 8 * half;
#pragma unroll
  for (int r = 0; r < 8; ++r) {
    const int k = 2 * (r & 3) + ((r >> 2) << 4);   // pair base per ISA layout
    f.d[r] = *(const unsigned*)(p + k);
  }
}

static __device__ inline v8f wmma_bf16(const FragBF& a, const FragBF& b, v8f c) {
  return __builtin_amdgcn_wmma_f32_16x16x32_bf16(false, a.v, false, b.v,
                                                 (short)0, c, false, false);
}

static __device__ inline v8f zero8() {
  v8f z = {0.f, 0.f, 0.f, 0.f, 0.f, 0.f, 0.f, 0.f};
  return z;
}

// ---------------------------------------------------------------------------
// Tensor Data Mover: 1D contiguous global -> LDS copy of n_u16 bf16 elements.
// D# packing per CDNA5 ISA (group0: count/lds/global/type,
// group1: data_size=2B, tensor_dim0=n, 1-row tile, stride=n).  Issued by one
// wave; completion via s_wait_tensorcnt then workgroup barrier.
// This toolchain's builtin is the 6-arg form:
//   (uint32x4 g0, int32x8 g1, int32x4 g2, int32x4 g3, int32x8 xtra, i32 cpol)
// ---------------------------------------------------------------------------
#if defined(__has_builtin)
#if __has_builtin(__builtin_amdgcn_tensor_load_to_lds)
#define HAVE_TDM 1
#endif
#endif

#ifdef HAVE_TDM
typedef __attribute__((ext_vector_type(4))) unsigned tdm_v4u;
typedef __attribute__((ext_vector_type(8))) int      tdm_v8i;
typedef __attribute__((ext_vector_type(4))) int      tdm_v4i;

static __device__ inline void tdm_load_lds_1d(unsigned lds_off, const void* gsrc,
                                              unsigned n_u16) {
  const unsigned long long ga = (unsigned long long)(size_t)gsrc;
  tdm_v4u g0;
  g0[0] = 1u;                                                  // count=1
  g0[1] = lds_off;                                             // lds_addr
  g0[2] = (unsigned)ga;                                        // global_addr lo
  g0[3] = (unsigned)((ga >> 32) & 0x01ffffffu) | (2u << 30);   // addr hi | type=2
  tdm_v8i g1;
  g1[0] = 1 << 16;                                             // data_size = 2B
  g1[1] = (int)((n_u16 & 0xffffu) << 16);                      // tensor_dim0 lo
  g1[2] = (int)(((n_u16 >> 16) & 0xffffu) | (1u << 16));       // dim0 hi | dim1=1
  g1[3] = (int)((n_u16 & 0xffffu) << 16);                      // tile_dim0
  g1[4] = 1;                                                   // tile_dim1=1
  g1[5] = (int)n_u16;                                          // dim0_stride lo
  g1[6] = 0;
  g1[7] = 0;
  const tdm_v4i gz = {0, 0, 0, 0};
  const tdm_v8i gz8 = {0, 0, 0, 0, 0, 0, 0, 0};
  __builtin_amdgcn_tensor_load_to_lds(g0, g1, gz, gz, gz8, 0);
}
#endif

// ---------------------------------------------------------------------------
// k0: convert weight matrices to bf16 (stored as ushort)
// ---------------------------------------------------------------------------
__global__ __launch_bounds__(256) void k0_convert(
    const float* wqkv, const float* wproj, const float* wmlp1, const float* wmlp2,
    unsigned short* oqkv, unsigned short* oproj,
    unsigned short* omlp1, unsigned short* omlp2) {
  const int i = blockIdx.x * 256 + threadIdx.x;
  if (i < 576 * 192) oqkv[i]  = f2bf(wqkv[i]);
  if (i < 192 * 192) oproj[i] = f2bf(wproj[i]);
  if (i < 768 * 192) omlp1[i] = f2bf(wmlp1[i]);
  if (i < 192 * 768) omlp2[i] = f2bf(wmlp2[i]);
}

// ---------------------------------------------------------------------------
// k1: window gather (with roll) + QKV GEMM.  One block per window.
//   Q bf16 [win][head][144][32]  (pre-scaled by hd^-0.5)
//   K bf16 [win][head][144][32]
//   Vt bf16 [win][head][32][144] (transposed for P@V B-fragments)
// ---------------------------------------------------------------------------
__global__ __launch_bounds__(256) void k1_qkv(
    const float* __restrict__ x, const unsigned short* __restrict__ wqkv,
    const float* __restrict__ bqkv, const int* __restrict__ rollp,
    unsigned short* __restrict__ Qb, unsigned short* __restrict__ Kb,
    unsigned short* __restrict__ Vt) {
  __shared__ __attribute__((aligned(16))) unsigned short sX[Twin * Cc]; // 55296 B
  const int roll = *rollp;
  const int win = blockIdx.x;
  const int iw = win >> 6, tw = win & 63, iz = tw >> 4, ih = tw & 15;
  const int tid = threadIdx.x;

  for (int idx = tid; idx < Twin * Cc; idx += 256) {
    const int t = idx / Cc, c = idx - t * Cc;
    const int tz = t / 72, rem = t - tz * 72, th = rem / 12, twd = rem - th * 12;
    int z = iz * 2 + tz, h = ih * 6 + th, w = iw * 12 + twd;
    if (roll) { z = (z + 1) & 7; h += 3; if (h >= Hc) h -= Hc; w += 6; if (w >= Wc) w -= Wc; }
    sX[idx] = f2bf(x[((size_t)(z * Hc + h) * Wc + w) * Cc + c]);
  }
  __syncthreads();

  const int wave = tid >> 5, lane = tid & 31, rn = lane & 15, half = lane >> 4;
  const float qscale = 0.17677669529663687f;   // 32^-0.5

  for (int tile = wave; tile < 9 * 36; tile += 8) {      // M=144/16, N=576/16
    const int tm = tile % 9, tn = tile / 9;
    const int m0 = tm * 16, n0 = tn * 16;
    v8f acc = zero8();
#pragma unroll
    for (int kk = 0; kk < Cc; kk += 32) {
      FragBF a, b;
      load_frag(a, sX, m0, Cc, kk);
      load_frag(b, wqkv, n0, Cc, kk);                    // wqkv[n][k], k contig
      acc = wmma_bf16(a, b, acc);
    }
    const int n = n0 + rn;
    const float bv = bqkv[n];
#pragma unroll
    for (int r = 0; r < 8; ++r) {
      const int m = m0 + r + 8 * half;
      const float v = acc[r] + bv;
      if (n < 192) {
        const int head = n >> 5, hd = n & 31;
        Qb[(((size_t)win * HEADS + head) * Twin + m) * HD + hd] = f2bf(v * qscale);
      } else if (n < 384) {
        const int n2 = n - 192, head = n2 >> 5, hd = n2 & 31;
        Kb[(((size_t)win * HEADS + head) * Twin + m) * HD + hd] = f2bf(v);
      } else {
        const int n2 = n - 384, head = n2 >> 5, hd = n2 & 31;
        Vt[(((size_t)win * HEADS + head) * HD + hd) * Twin + m] = f2bf(v);
      }
    }
  }
}

// ---------------------------------------------------------------------------
// k2: attention per (window, head). Register-resident 16x144 score strip.
// ---------------------------------------------------------------------------
__global__ __launch_bounds__(256) void k2_attn(
    const unsigned short* __restrict__ Qb, const unsigned short* __restrict__ Kb,
    const unsigned short* __restrict__ Vt, const float* __restrict__ bias_table,
    const int* __restrict__ rollp, unsigned short* __restrict__ Xattn) {
  __shared__ __attribute__((aligned(16))) unsigned short sK[Twin * HD];   //  9216 B
  __shared__ __attribute__((aligned(16))) unsigned short sV[HD * 160];    // 10240 B (padded)
  __shared__ __attribute__((aligned(16))) unsigned short sP[8 * 16 * 160];// 40960 B
  const int roll = *rollp;
  const int wh_ = blockIdx.x;
  const int win = wh_ / HEADS, head = wh_ - win * HEADS;
  const int tw = win & 63, iz = tw >> 4, ih = tw & 15;
  const int tid = threadIdx.x;

  // stage K slab (contiguous 9216 B) via the Tensor Data Mover
#ifdef HAVE_TDM
  if (tid == 0) {
    tdm_load_lds_1d((unsigned)(size_t)(void*)sK,
                    Kb + (size_t)wh_ * (Twin * HD), Twin * HD);
  }
#else
  {
    const unsigned* ksrc = (const unsigned*)(Kb + (size_t)wh_ * (Twin * HD));
    unsigned* kdst = (unsigned*)sK;
    for (int i = tid; i < (Twin * HD) / 2; i += 256) kdst[i] = ksrc[i];
  }
#endif
  { // stage V (layout change: pad K-dim 144 -> 160), manual
    const unsigned short* vsrc = Vt + (size_t)wh_ * (Twin * HD);
    for (int i = tid; i < HD * Twin; i += 256) {
      const int hd = i / Twin, t = i - hd * Twin;
      sV[hd * 160 + t] = vsrc[i];
    }
    for (int i = tid; i < HD * 16; i += 256) sV[(i >> 4) * 160 + Twin + (i & 15)] = 0;
  }
#ifdef HAVE_TDM
  if (tid == 0) __builtin_amdgcn_s_wait_tensorcnt(0);
#endif
  __syncthreads();

  const int wave = tid >> 5, lane = tid & 31, rn = lane & 15, half = lane >> 4;
  unsigned short* strip = sP + wave * 16 * 160;
  const unsigned short* qbase = Qb + (size_t)wh_ * (Twin * HD);

  for (int qt = wave; qt < 9; qt += 8) {
    const int m0 = qt * 16;
    FragBF a;
    load_frag(a, qbase, m0, HD, 0);
    v8f acc[9];
#pragma unroll
    for (int j = 0; j < 9; ++j) {                    // S = Q @ K^T, K-dim = 32
      FragBF b;
      load_frag(b, sK, j * 16, HD, 0);
      acc[j] = wmma_bf16(a, b, zero8());
    }
    // earth-specific bias + shift mask, in-register
#pragma unroll
    for (int j = 0; j < 9; ++j) {
      const int nt = j * 16 + rn;                    // key token
      const int tzk = nt / 72, remk = nt - tzk * 72;
      const int thk = remk / 12, twk = remk - thk * 12;
      int labk = 0;
      if (roll) {
        const int zz = iz * 2 + tzk, hh = ih * 6 + thk;
        labk = ((zz < 6) ? 0 : ((zz < 7) ? 1 : 2)) * 3 +
               ((hh < 90) ? 0 : ((hh < 93) ? 1 : 2));
      }
#pragma unroll
      for (int r = 0; r < 8; ++r) {
        const int mt = m0 + r + 8 * half;            // query token
        const int tzq = mt / 72, remq = mt - tzq * 72;
        const int thq = remq / 12, twq = remq - thq * 12;
        const int pos = (tzq + 2 * tzk) * 828 + (thq + 6 * thk) * 23 + (twq - twk + 11);
        float bm = bias_table[(size_t)pos * (TWc * HEADS) + tw * HEADS + head];
        if (roll) {
          const int zz = iz * 2 + tzq, hh = ih * 6 + thq;
          const int labq = ((zz < 6) ? 0 : ((zz < 7) ? 1 : 2)) * 3 +
                           ((hh < 90) ? 0 : ((hh < 93) ? 1 : 2));
          if (labq != labk) bm -= 100.0f;
        }
        acc[j][r] += bm;
      }
    }
    // softmax across the 144 keys; rows live in 16-lane halves (wave32)
    float inv_s[8];
#pragma unroll
    for (int r = 0; r < 8; ++r) {
      float mx = acc[0][r];
#pragma unroll
      for (int j = 1; j < 9; ++j) mx = fmaxf(mx, acc[j][r]);
      for (int d = 1; d < 16; d <<= 1) mx = fmaxf(mx, __shfl_xor(mx, d, 32));
      float s = 0.f;
#pragma unroll
      for (int j = 0; j < 9; ++j) { const float e = __expf(acc[j][r] - mx); acc[j][r] = e; s += e; }
      for (int d = 1; d < 16; d <<= 1) s += __shfl_xor(s, d, 32);
      inv_s[r] = 1.0f / s;
    }
    // stage unnormalized P (bf16) into this wave's strip, zero the K padding
#pragma unroll
    for (int r = 0; r < 8; ++r) {
      const int rowl = r + 8 * half;
#pragma unroll
      for (int j = 0; j < 9; ++j) strip[rowl * 160 + j * 16 + rn] = f2bf(acc[j][r]);
      strip[rowl * 160 + Twin + rn] = 0;
    }
    // O = P @ V  (K padded to 160 -> 5 WMMA steps), normalize at write
#pragma unroll
    for (int nt2 = 0; nt2 < 2; ++nt2) {
      v8f o = zero8();
#pragma unroll
      for (int kk = 0; kk < 160; kk += 32) {
        FragBF a2, b2;
        load_frag(a2, strip, 0, 160, kk);
        load_frag(b2, sV, nt2 * 16, 160, kk);
        o = wmma_bf16(a2, b2, o);
      }
      const int hd = nt2 * 16 + rn;
#pragma unroll
      for (int r = 0; r < 8; ++r) {
        const int m = m0 + r + 8 * half;
        Xattn[((size_t)win * Twin + m) * Cc + head * HD + hd] = f2bf(o[r] * inv_s[r]);
      }
    }
  }
}

// ---------------------------------------------------------------------------
// k3a: proj GEMM per window -> Po (f32, window-token order)
// ---------------------------------------------------------------------------
__global__ __launch_bounds__(256) void k3a_proj(
    const unsigned short* __restrict__ Xattn, const unsigned short* __restrict__ wproj,
    const float* __restrict__ bproj, float* __restrict__ Po) {
  __shared__ __attribute__((aligned(16))) unsigned short sX[Twin * Cc]; // 55296 B
  const int win = blockIdx.x, tid = threadIdx.x;
#ifdef HAVE_TDM
  if (tid == 0) {
    tdm_load_lds_1d((unsigned)(size_t)(void*)sX,
                    Xattn + (size_t)win * Twin * Cc, Twin * Cc);
    __builtin_amdgcn_s_wait_tensorcnt(0);
  }
#else
  {
    const unsigned* src = (const unsigned*)(Xattn + (size_t)win * Twin * Cc);
    unsigned* dst = (unsigned*)sX;
    for (int i = tid; i < (Twin * Cc) / 2; i += 256) dst[i] = src[i];
  }
#endif
  __syncthreads();
  const int wave = tid >> 5, lane = tid & 31, rn = lane & 15, half = lane >> 4;
  for (int tile = wave; tile < 9 * 12; tile += 8) {
    const int tm = tile % 9, tn = tile / 9;
    const int m0 = tm * 16, n0 = tn * 16;
    v8f acc = zero8();
#pragma unroll
    for (int kk = 0; kk < Cc; kk += 32) {
      FragBF a, b;
      load_frag(a, sX, m0, Cc, kk);
      load_frag(b, wproj, n0, Cc, kk);
      acc = wmma_bf16(a, b, acc);
    }
    const int n = n0 + rn;
    const float bv = bproj[n];
#pragma unroll
    for (int r = 0; r < 8; ++r) {
      const int m = m0 + r + 8 * half;
      Po[((size_t)win * Twin + m) * Cc + n] = acc[r] + bv;
    }
  }
}

// ---------------------------------------------------------------------------
// k3b: window reverse + unroll + LN1 + residual.  Thread = window-token.
// ---------------------------------------------------------------------------
__global__ __launch_bounds__(256) void k3b_lnres(
    const float* __restrict__ Po, const float* __restrict__ x,
    const float* __restrict__ g1, const float* __restrict__ beta1,
    const int* __restrict__ rollp, float* __restrict__ Yf,
    unsigned short* __restrict__ Yb) {
  const int i = blockIdx.x * 256 + threadIdx.x;     // < NT, window-token index
  const int roll = *rollp;
  const int win = i / Twin, t = i - win * Twin;
  const int iw = win >> 6, tw = win & 63, iz = tw >> 4, ih = tw & 15;
  const int tz = t / 72, rem = t - tz * 72, th = rem / 12, twd = rem - th * 12;
  int z = iz * 2 + tz, h = ih * 6 + th, w = iw * 12 + twd;
  if (roll) { z = (z + 1) & 7; h += 3; if (h >= Hc) h -= Hc; w += 6; if (w >= Wc) w -= Wc; }
  const size_t tok = (size_t)(z * Hc + h) * Wc + w;
  const float* po = Po + (size_t)i * Cc;
  float s = 0.f, s2 = 0.f;
  for (int c = 0; c < Cc; ++c) { const float v = po[c]; s += v; s2 += v * v; }
  const float mean = s * (1.0f / Cc);
  const float var = s2 * (1.0f / Cc) - mean * mean;
  const float rs = rsqrtf(var + 1e-5f);
  const float* xr = x + tok * Cc;
  float* yf = Yf + tok * Cc;
  unsigned short* yb = Yb + tok * Cc;
  for (int c = 0; c < Cc; ++c) {
    const float ln = (po[c] - mean) * rs * g1[c] + beta1[c];
    const float yv = xr[c] + ln;
    yf[c] = yv;
    yb[c] = f2bf(yv);
  }
}

// ---------------------------------------------------------------------------
// k4a: MLP1 + exact GELU.  4 N-tiles per wave (A-fragment reuse).
// ---------------------------------------------------------------------------
__global__ __launch_bounds__(256) void k4a_mlp1(
    const unsigned short* __restrict__ Yb, const unsigned short* __restrict__ w1,
    const float* __restrict__ b1, unsigned short* __restrict__ Hb) {
  const int gw = blockIdx.x * 8 + (threadIdx.x >> 5);   // global wave id
  const int tm = gw % (NT / 16), tng = gw / (NT / 16);  // tng < 12 (groups of 4)
  const int m0 = tm * 16, nbase = tng * 64;
  const int lane = threadIdx.x & 31, rn = lane & 15, half = lane >> 4;
  __builtin_prefetch(Yb + (size_t)m0 * Cc, 0, 1);
  __builtin_prefetch(w1 + (size_t)nbase * Cc, 0, 1);
  v8f acc[4] = {zero8(), zero8(), zero8(), zero8()};
#pragma unroll
  for (int kk = 0; kk < Cc; kk += 32) {
    FragBF a;
    load_frag(a, Yb, m0, Cc, kk);
#pragma unroll
    for (int t = 0; t < 4; ++t) {
      FragBF b;
      load_frag(b, w1, nbase + t * 16, Cc, kk);
      acc[t] = wmma_bf16(a, b, acc[t]);
    }
  }
#pragma unroll
  for (int t = 0; t < 4; ++t) {
    const int n = nbase + t * 16 + rn;
    const float bv = b1[n];
#pragma unroll
    for (int r = 0; r < 8; ++r) {
      const int m = m0 + r + 8 * half;
      const float v = acc[t][r] + bv;
      const float g = 0.5f * v * (1.0f + erff(v * 0.70710678118654752f));
      Hb[(size_t)m * 768 + n] = f2bf(g);
    }
  }
}

// ---------------------------------------------------------------------------
// k4b: MLP2.  4 N-tiles per wave, K = 768.
// ---------------------------------------------------------------------------
__global__ __launch_bounds__(256) void k4b_mlp2(
    const unsigned short* __restrict__ Hb, const unsigned short* __restrict__ w2,
    const float* __restrict__ b2, float* __restrict__ Po) {
  const int gw = blockIdx.x * 8 + (threadIdx.x >> 5);
  const int tm = gw % (NT / 16), tng = gw / (NT / 16);  // tng < 3 (groups of 4)
  const int m0 = tm * 16, nbase = tng * 64;
  const int lane = threadIdx.x & 31, rn = lane & 15, half = lane >> 4;
  __builtin_prefetch(Hb + (size_t)m0 * 768, 0, 1);
  __builtin_prefetch(w2 + (size_t)nbase * 768, 0, 1);
  v8f acc[4] = {zero8(), zero8(), zero8(), zero8()};
#pragma unroll 4
  for (int kk = 0; kk < 768; kk += 32) {
    FragBF a;
    load_frag(a, Hb, m0, 768, kk);
#pragma unroll
    for (int t = 0; t < 4; ++t) {
      FragBF b;
      load_frag(b, w2, nbase + t * 16, 768, kk);
      acc[t] = wmma_bf16(a, b, acc[t]);
    }
  }
#pragma unroll
  for (int t = 0; t < 4; ++t) {
    const int n = nbase + t * 16 + rn;
    const float bv = b2[n];
#pragma unroll
    for (int r = 0; r < 8; ++r) {
      const int m = m0 + r + 8 * half;
      Po[(size_t)m * Cc + n] = acc[t][r] + bv;
    }
  }
}

// ---------------------------------------------------------------------------
// k4c: LN2 + residual -> final output (f32).  Thread = token (plain order).
// ---------------------------------------------------------------------------
__global__ __launch_bounds__(256) void k4c_out(
    const float* __restrict__ Po, const float* __restrict__ Yf,
    const float* __restrict__ g2, const float* __restrict__ beta2,
    float* __restrict__ out) {
  const int i = blockIdx.x * 256 + threadIdx.x;   // token
  const float* po = Po + (size_t)i * Cc;
  const float* yf = Yf + (size_t)i * Cc;
  float s = 0.f, s2 = 0.f;
  for (int c = 0; c < Cc; ++c) { const float v = po[c]; s += v; s2 += v * v; }
  const float mean = s * (1.0f / Cc);
  const float var = s2 * (1.0f / Cc) - mean * mean;
  const float rs = rsqrtf(var + 1e-5f);
  float* o = out + (size_t)i * Cc;
  for (int c = 0; c < Cc; ++c)
    o[c] = yf[c] + (po[c] - mean) * rs * g2[c] + beta2[c];
}

// ---------------------------------------------------------------------------
// Workspace layout (bytes, all offsets 256-aligned). Total ~737 MB.
// ---------------------------------------------------------------------------
static constexpr size_t O_WQKV  = 0;                       // 576*192*2
static constexpr size_t O_WPROJ = 221184;                  // 192*192*2
static constexpr size_t O_WMLP1 = 294912;                  // 768*192*2
static constexpr size_t O_WMLP2 = 589824;                  // 192*768*2
static constexpr size_t O_QB    = 884736;                  // WIN*6*144*32*2
static constexpr size_t O_KB    = O_QB    + 56623104;
static constexpr size_t O_VT    = O_KB    + 56623104;
static constexpr size_t O_XATTN = O_VT    + 56623104;
static constexpr size_t O_PO    = O_XATTN + 56623104;      // NT*192*4 (reused)
static constexpr size_t O_YF    = O_PO    + 113246208;     // NT*192*4
static constexpr size_t O_YB    = O_YF    + 113246208;     // NT*192*2
static constexpr size_t O_HB    = O_YB    + 56623104;      // NT*768*2

extern "C" void kernel_launch(void* const* d_in, const int* in_sizes, int n_in,
                              void* d_out, int out_size, void* d_ws, size_t ws_size,
                              hipStream_t stream) {
  (void)in_sizes; (void)n_in; (void)out_size; (void)ws_size;
  const float* x          = (const float*)d_in[0];
  const float* w_qkv      = (const float*)d_in[1];
  const float* b_qkv      = (const float*)d_in[2];
  const float* w_proj     = (const float*)d_in[3];
  const float* b_proj     = (const float*)d_in[4];
  const float* bias_table = (const float*)d_in[5];
  const float* g1         = (const float*)d_in[6];
  const float* beta1      = (const float*)d_in[7];
  const float* g2         = (const float*)d_in[8];
  const float* beta2      = (const float*)d_in[9];
  const float* w_mlp1     = (const float*)d_in[10];
  const float* b_mlp1     = (const float*)d_in[11];
  const float* w_mlp2     = (const float*)d_in[12];
  const float* b_mlp2     = (const float*)d_in[13];
  const int*   roll       = (const int*)d_in[14];

  char* ws = (char*)d_ws;
  unsigned short* wqkv_b  = (unsigned short*)(ws + O_WQKV);
  unsigned short* wproj_b = (unsigned short*)(ws + O_WPROJ);
  unsigned short* wmlp1_b = (unsigned short*)(ws + O_WMLP1);
  unsigned short* wmlp2_b = (unsigned short*)(ws + O_WMLP2);
  unsigned short* Qb      = (unsigned short*)(ws + O_QB);
  unsigned short* Kb      = (unsigned short*)(ws + O_KB);
  unsigned short* Vt      = (unsigned short*)(ws + O_VT);
  unsigned short* Xattn   = (unsigned short*)(ws + O_XATTN);
  float*          Po      = (float*)(ws + O_PO);
  float*          Yf      = (float*)(ws + O_YF);
  unsigned short* Yb      = (unsigned short*)(ws + O_YB);
  unsigned short* Hb      = (unsigned short*)(ws + O_HB);
  float*          out     = (float*)d_out;

  k0_convert<<<NT / 256, 256, 0, stream>>>(w_qkv, w_proj, w_mlp1, w_mlp2,
                                           wqkv_b, wproj_b, wmlp1_b, wmlp2_b);
  k1_qkv<<<WIN, 256, 0, stream>>>(x, wqkv_b, b_qkv, roll, Qb, Kb, Vt);
  k2_attn<<<WIN * HEADS, 256, 0, stream>>>(Qb, Kb, Vt, bias_table, roll, Xattn);
  k3a_proj<<<WIN, 256, 0, stream>>>(Xattn, wproj_b, b_proj, Po);
  k3b_lnres<<<NT / 256, 256, 0, stream>>>(Po, x, g1, beta1, roll, Yf, Yb);
  k4a_mlp1<<<(NT / 16) * 12 / 8, 256, 0, stream>>>(Yb, wmlp1_b, b_mlp1, Hb);
  k4b_mlp2<<<(NT / 16) * 3 / 8, 256, 0, stream>>>(Hb, wmlp2_b, b_mlp2, Po);
  k4c_out<<<NT / 256, 256, 0, stream>>>(Po, Yf, g2, beta2, out);
}
